// CMTF_65601330479500
// MI455X (gfx1250) — matrix operand order repaired and checked by reading, not compiled, not run on hardware
//
#include <hip/hip_runtime.h>

typedef __bf16 bf16_t;
typedef __attribute__((ext_vector_type(8)))  float  v8f;
typedef __attribute__((ext_vector_type(8)))  __bf16 v8bf;
typedef __attribute__((ext_vector_type(16))) __bf16 v16bf;

#define MTOK 16384   // B*NQ = 256*64 tokens
#define DD   512

// ---- async global->LDS copy path (CDNA5 GLOBAL_LOAD_ASYNC_TO_LDS_B128) ----
#if defined(__has_builtin)
#if __has_builtin(__builtin_amdgcn_global_load_async_to_lds_b128) && \
    __has_builtin(__builtin_amdgcn_s_wait_asynccnt)
#define USE_ASYNC_LDS 1
#endif
#endif

#ifdef USE_ASYNC_LDS
typedef __attribute__((ext_vector_type(4))) int v4i;
typedef __attribute__((address_space(1))) v4i v4i_g;   // global ("__device__")
typedef __attribute__((address_space(3))) v4i v4i_l;   // LDS ("__shared__")
#define ASYNC_CP16(dst_lds, src_glob) \
  __builtin_amdgcn_global_load_async_to_lds_b128( \
      (v4i_g*)(src_glob), (v4i_l*)(dst_lds), 0, 0)
#define ASYNC_WAIT(n) __builtin_amdgcn_s_wait_asynccnt(n)
#endif

static __device__ __forceinline__ bf16_t f2bf(float f) {
  unsigned u = __builtin_bit_cast(unsigned, f);
  u += 0x7FFFu + ((u >> 16) & 1u);                 // round-to-nearest-even
  unsigned short s = (unsigned short)(u >> 16);
  return __builtin_bit_cast(bf16_t, s);
}
static __device__ __forceinline__ float bf2f(bf16_t b) {
  unsigned short s = __builtin_bit_cast(unsigned short, b);
  unsigned u = ((unsigned)s) << 16;
  return __builtin_bit_cast(float, u);
}
static __device__ __forceinline__ v8bf ld8(const bf16_t* p) { return *(const v8bf*)p; }
static __device__ __forceinline__ v16bf cat16(v8bf lo, v8bf hi) {
  return __builtin_shufflevector(lo, hi, 0,1,2,3,4,5,6,7,8,9,10,11,12,13,14,15);
}
static __device__ __forceinline__ v8f v8f_zero() {
  v8f z = {0.f,0.f,0.f,0.f,0.f,0.f,0.f,0.f};
  return z;
}
#define WMMA_BF16(a,b,c) \
  __builtin_amdgcn_wmma_f32_16x16x32_bf16(false,(a),false,(b),(short)0,(c),false,false)

#define TILE_ELEMS (128*40)   // one LDS tile buffer (128 rows, stride 40 bf16)

// ---------------------------------------------------------------------------
// Generic NT GEMM:  C[M,N] = A[M,K](row, lda) * Bt[N,K](row) + bias
// flags: bit0 = bf16 output, bit1 = leaky_relu(0.01)
// grid = (N/128, M/128), block = 256 (8 wave32 waves); wave tile 32x64
// Double-buffered LDS tiles staged via async global->LDS copies when available.
// ---------------------------------------------------------------------------
__global__ __launch_bounds__(256) void gemm_nt(
    const bf16_t* __restrict__ A, int lda,
    const bf16_t* __restrict__ Bt,
    const float* __restrict__ bias,
    void* __restrict__ Cp, int N, int K, int flags)
{
  __shared__ __align__(16) bf16_t As[2*TILE_ELEMS];
  __shared__ __align__(16) bf16_t Bs[2*TILE_ELEMS];
  const int t    = threadIdx.x;
  const int m0   = blockIdx.y * 128;
  const int n0   = blockIdx.x * 128;
  const int lrow = t >> 1;
  const int lcol = (t & 1) << 4;
  const int wid  = t >> 5, lane = t & 31;
  const int wm   = wid & 3, wn = wid >> 2;
  const int g    = lane >> 4, l15 = lane & 15;

  v8f acc[2][4];
  #pragma unroll
  for (int i = 0; i < 2; ++i)
    #pragma unroll
    for (int j = 0; j < 4; ++j) acc[i][j] = v8f_zero();

  const bf16_t* agBase = A  + (size_t)(m0 + lrow) * lda + lcol;
  const bf16_t* bgBase = Bt + (size_t)(n0 + lrow) * K   + lcol;
  bf16_t* aDst = &As[lrow*40 + lcol];
  bf16_t* bDst = &Bs[lrow*40 + lcol];

#ifdef USE_ASYNC_LDS
  const int nIter = K >> 5;
  // preload tile 0 into buffer 0 (4 async b128 ops per wave)
  ASYNC_CP16(aDst,     agBase);
  ASYNC_CP16(aDst + 8, agBase + 8);
  ASYNC_CP16(bDst,     bgBase);
  ASYNC_CP16(bDst + 8, bgBase + 8);
  for (int it = 0; it < nIter; ++it) {
    const int cur = it & 1;
    if (it + 1 < nIter) {            // stage next tile into the other buffer
      const int ko = (it + 1) << 5;
      const int lo = (cur ^ 1) * TILE_ELEMS;
      ASYNC_CP16(aDst + lo,     agBase + ko);
      ASYNC_CP16(aDst + lo + 8, agBase + ko + 8);
      ASYNC_CP16(bDst + lo,     bgBase + ko);
      ASYNC_CP16(bDst + lo + 8, bgBase + ko + 8);
      ASYNC_WAIT(4);                 // in-order: current tile's 4 ops are done
    } else {
      ASYNC_WAIT(0);
    }
    __syncthreads();                 // all waves' current-tile data visible
    const int so = cur * TILE_ELEMS;
    v16bf af[2], bfr[4];
    #pragma unroll
    for (int i = 0; i < 2; ++i) {
      int r = wm*32 + i*16 + l15;
      af[i] = cat16(ld8(&As[so + r*40 + 8*g]), ld8(&As[so + r*40 + 16 + 8*g]));
    }
    #pragma unroll
    for (int j = 0; j < 4; ++j) {
      int r = wn*64 + j*16 + l15;
      bfr[j] = cat16(ld8(&Bs[so + r*40 + 8*g]), ld8(&Bs[so + r*40 + 16 + 8*g]));
    }
    #pragma unroll
    for (int i = 0; i < 2; ++i)
      #pragma unroll
      for (int j = 0; j < 4; ++j)
        acc[i][j] = WMMA_BF16(af[i], bfr[j], acc[i][j]);
    __syncthreads();                 // everyone done reading before restaging
  }
#else
  for (int k0 = 0; k0 < K; k0 += 32) {
    const uint4* ag = (const uint4*)(agBase + k0);
    const uint4* bg = (const uint4*)(bgBase + k0);
    uint4 a0 = ag[0], a1 = ag[1];
    uint4 b0 = bg[0], b1 = bg[1];
    if (k0 + 32 < K) {
      __builtin_prefetch(agBase + k0 + 32, 0, 1);
      __builtin_prefetch(bgBase + k0 + 32, 0, 1);
    }
    uint4* ad = (uint4*)aDst;
    uint4* bd = (uint4*)bDst;
    ad[0] = a0; ad[1] = a1; bd[0] = b0; bd[1] = b1;
    __syncthreads();
    v16bf af[2], bfr[4];
    #pragma unroll
    for (int i = 0; i < 2; ++i) {
      int r = wm*32 + i*16 + l15;
      af[i] = cat16(ld8(&As[r*40 + 8*g]), ld8(&As[r*40 + 16 + 8*g]));
    }
    #pragma unroll
    for (int j = 0; j < 4; ++j) {
      int r = wn*64 + j*16 + l15;
      bfr[j] = cat16(ld8(&Bs[r*40 + 8*g]), ld8(&Bs[r*40 + 16 + 8*g]));
    }
    #pragma unroll
    for (int i = 0; i < 2; ++i)
      #pragma unroll
      for (int j = 0; j < 4; ++j)
        acc[i][j] = WMMA_BF16(af[i], bfr[j], acc[i][j]);
    __syncthreads();
  }
#endif

  #pragma unroll
  for (int i = 0; i < 2; ++i) {
    const int mb = m0 + wm*32 + i*16 + 8*g;
    #pragma unroll
    for (int j = 0; j < 4; ++j) {
      const int nn = n0 + wn*64 + j*16 + l15;
      const float bv = bias[nn];
      #pragma unroll
      for (int r = 0; r < 8; ++r) {
        float v = acc[i][j][r] + bv;
        if (flags & 2) v = (v > 0.f) ? v : 0.01f * v;
        if (flags & 1) ((bf16_t*)Cp)[(size_t)(mb + r) * N + nn] = f2bf(v);
        else           ((float*)Cp) [(size_t)(mb + r) * N + nn] = v;
      }
    }
  }
}

// ---------------------------------------------------------------------------
// Implicit-im2col 3x3 SAME conv as GEMM: M=16384 tokens(b,y,x), K=9*512, N=512
// S:[16384,512] bf16 token-major conv input; Wt:[512,4608] with k = tap*512+c
// Every wave always has >=1 valid lane per tap (16 consecutive tokens span two
// distinct image rows, x covers 0..7), so async-op counts stay wave-uniform.
// ---------------------------------------------------------------------------
__global__ __launch_bounds__(256) void conv_gemm(
    const bf16_t* __restrict__ S, const bf16_t* __restrict__ Wt,
    const float* __restrict__ bias, bf16_t* __restrict__ C)
{
  __shared__ __align__(16) bf16_t As[2*TILE_ELEMS];
  __shared__ __align__(16) bf16_t Bs[2*TILE_ELEMS];
  const int t    = threadIdx.x;
  const int m0   = blockIdx.y * 128;
  const int n0   = blockIdx.x * 128;
  const int lrow = t >> 1;
  const int lcol = (t & 1) << 4;
  const int wid  = t >> 5, lane = t & 31;
  const int wm   = wid & 3, wn = wid >> 2;
  const int g    = lane >> 4, l15 = lane & 15;

  const int token = m0 + lrow;
  const int bb = token >> 6, pos = token & 63;
  const int y = pos >> 3, x = pos & 7;
  bf16_t* aDst = &As[lrow*40 + lcol];
  bf16_t* bDst = &Bs[lrow*40 + lcol];
  const bf16_t* bgBase = Wt + (size_t)(n0 + lrow) * 4608 + lcol;

  v8f acc[2][4];
  #pragma unroll
  for (int i = 0; i < 2; ++i)
    #pragma unroll
    for (int j = 0; j < 4; ++j) acc[i][j] = v8f_zero();

#ifdef USE_ASYNC_LDS
  const int nIter = 4608 >> 5;       // 144
  {
    const int dy = -1, dx = -1;      // tap 0
    const int yy = y + dy, xx = x + dx;
    if ((unsigned)yy < 8u && (unsigned)xx < 8u) {
      const bf16_t* src = S + ((size_t)((bb<<6)+(yy<<3)+xx))*DD + lcol;
      ASYNC_CP16(aDst,     src);
      ASYNC_CP16(aDst + 8, src + 8);
    } else {
      uint4 z; z.x = z.y = z.z = z.w = 0u;
      ((uint4*)aDst)[0] = z; ((uint4*)aDst)[1] = z;
    }
    ASYNC_CP16(bDst,     bgBase);
    ASYNC_CP16(bDst + 8, bgBase + 8);
  }
  for (int it = 0; it < nIter; ++it) {
    const int cur = it & 1;
    if (it + 1 < nIter) {
      const int ko  = (it + 1) << 5;
      const int tap = ko >> 9;
      const int dy  = tap / 3 - 1;
      const int dx  = tap - (tap / 3) * 3 - 1;
      const int yy = y + dy, xx = x + dx;
      const int lo = (cur ^ 1) * TILE_ELEMS;
      if ((unsigned)yy < 8u && (unsigned)xx < 8u) {
        const bf16_t* src = S + ((size_t)((bb<<6)+(yy<<3)+xx))*DD + (ko & 511) + lcol;
        ASYNC_CP16(aDst + lo,     src);
        ASYNC_CP16(aDst + lo + 8, src + 8);
      } else {
        uint4 z; z.x = z.y = z.z = z.w = 0u;
        ((uint4*)(aDst + lo))[0] = z; ((uint4*)(aDst + lo))[1] = z;
      }
      ASYNC_CP16(bDst + lo,     bgBase + ko);
      ASYNC_CP16(bDst + lo + 8, bgBase + ko + 8);
      ASYNC_WAIT(4);
    } else {
      ASYNC_WAIT(0);
    }
    __syncthreads();
    const int so = cur * TILE_ELEMS;
    v16bf af[2], bfr[4];
    #pragma unroll
    for (int i = 0; i < 2; ++i) {
      int r = wm*32 + i*16 + l15;
      af[i] = cat16(ld8(&As[so + r*40 + 8*g]), ld8(&As[so + r*40 + 16 + 8*g]));
    }
    #pragma unroll
    for (int j = 0; j < 4; ++j) {
      int r = wn*64 + j*16 + l15;
      bfr[j] = cat16(ld8(&Bs[so + r*40 + 8*g]), ld8(&Bs[so + r*40 + 16 + 8*g]));
    }
    #pragma unroll
    for (int i = 0; i < 2; ++i)
      #pragma unroll
      for (int j = 0; j < 4; ++j)
        acc[i][j] = WMMA_BF16(af[i], bfr[j], acc[i][j]);
    __syncthreads();
  }
#else
  for (int k0 = 0; k0 < 4608; k0 += 32) {
    const int tap = k0 >> 9;
    const int dy = tap / 3 - 1;
    const int dx = tap - (tap / 3) * 3 - 1;
    const int yy = y + dy, xx = x + dx;
    uint4* ad = (uint4*)aDst;
    if ((unsigned)yy < 8u && (unsigned)xx < 8u) {
      const uint4* ag = (const uint4*)(S + ((size_t)((bb<<6)+(yy<<3)+xx))*DD + (k0 & 511) + lcol);
      ad[0] = ag[0]; ad[1] = ag[1];
    } else {
      uint4 z; z.x = z.y = z.z = z.w = 0u;
      ad[0] = z; ad[1] = z;
    }
    const uint4* bg = (const uint4*)(bgBase + k0);
    uint4* bd = (uint4*)bDst;
    bd[0] = bg[0]; bd[1] = bg[1];
    __syncthreads();
    v16bf af[2], bfr[4];
    #pragma unroll
    for (int i = 0; i < 2; ++i) {
      int r = wm*32 + i*16 + l15;
      af[i] = cat16(ld8(&As[r*40 + 8*g]), ld8(&As[r*40 + 16 + 8*g]));
    }
    #pragma unroll
    for (int j = 0; j < 4; ++j) {
      int r = wn*64 + j*16 + l15;
      bfr[j] = cat16(ld8(&Bs[r*40 + 8*g]), ld8(&Bs[r*40 + 16 + 8*g]));
    }
    #pragma unroll
    for (int i = 0; i < 2; ++i)
      #pragma unroll
      for (int j = 0; j < 4; ++j)
        acc[i][j] = WMMA_BF16(af[i], bfr[j], acc[i][j]);
    __syncthreads();
  }
#endif

  #pragma unroll
  for (int i = 0; i < 2; ++i) {
    const int mb = m0 + wm*32 + i*16 + 8*g;
    #pragma unroll
    for (int j = 0; j < 4; ++j) {
      const int nn = n0 + wn*64 + j*16 + l15;
      const float bv = bias[nn];
      #pragma unroll
      for (int r = 0; r < 8; ++r)
        C[(size_t)(mb + r) * DD + nn] = f2bf(acc[i][j][r] + bv);
    }
  }
}

// ---------------------------------------------------------------------------
// Attention for one (batch, head): 64 queries/keys, DK=64.
// scores = Q·Kt * 0.125, softmax rows, out = P·V.  All matmuls via WMMA.
// grid = B*H = 2048, block = 256.  Q/K/V/O are [16384,512] bf16, head h cols.
// ---------------------------------------------------------------------------
__global__ __launch_bounds__(256) void attn_kernel(
    const bf16_t* __restrict__ Q, const bf16_t* __restrict__ Km,
    const bf16_t* __restrict__ V, bf16_t* __restrict__ O)
{
  __shared__ __align__(16) bf16_t Qs[64*72];
  __shared__ __align__(16) bf16_t Ks[64*72];
  __shared__ __align__(16) bf16_t Vts[64*72];
  __shared__ __align__(16) bf16_t Ps[64*72];
  __shared__ float Ss[64*64];

  const int t = threadIdx.x;
  const int b = blockIdx.x >> 3;
  const int h = blockIdx.x & 7;
  const size_t hoff = (size_t)h * 64;

  { // cooperative loads; V is stored transposed (Vts[d][m])
    const int row = t >> 2;
    const int cb  = (t & 3) << 4;
    const size_t gbase = ((size_t)(b*64 + row)) * DD + hoff + cb;
    const uint4* qg = (const uint4*)(Q  + gbase);
    const uint4* kg = (const uint4*)(Km + gbase);
    uint4* qd = (uint4*)&Qs[row*72 + cb];
    uint4* kd = (uint4*)&Ks[row*72 + cb];
    qd[0] = qg[0]; qd[1] = qg[1];
    kd[0] = kg[0]; kd[1] = kg[1];
    const bf16_t* vg = V + gbase;
    #pragma unroll
    for (int i = 0; i < 16; ++i) Vts[(cb + i)*72 + row] = vg[i];
  }
  __syncthreads();

  const int wid = t >> 5, lane = t & 31;
  const int g = lane >> 4, l15 = lane & 15;
  const int ntile = wid & 3;          // 16-row query block
  const int pbase = (wid >> 2) << 1;  // two 16-col blocks per wave
  const int qrow  = ntile*16 + l15;

  // ---- scores = Q @ K^T  (K stored row-major == NT layout) ----
  v8f s0 = v8f_zero(), s1 = v8f_zero();
  #pragma unroll
  for (int s = 0; s < 2; ++s) {
    v16bf af = cat16(ld8(&Qs[qrow*72 + s*32 + 16*g]),
                     ld8(&Qs[qrow*72 + s*32 + 16*g + 8]));
    int k0r = (pbase + 0)*16 + l15;
    int k1r = (pbase + 1)*16 + l15;
    v16bf b0 = cat16(ld8(&Ks[k0r*72 + s*32 + 16*g]), ld8(&Ks[k0r*72 + s*32 + 16*g + 8]));
    v16bf b1 = cat16(ld8(&Ks[k1r*72 + s*32 + 16*g]), ld8(&Ks[k1r*72 + s*32 + 16*g + 8]));
    s0 = WMMA_BF16(af, b0, s0);
    s1 = WMMA_BF16(af, b1, s1);
  }
  #pragma unroll
  for (int r = 0; r < 8; ++r) {
    Ss[(ntile*16 + 8*g + r)*64 + (pbase + 0)*16 + l15] = s0[r] * 0.125f;
    Ss[(ntile*16 + 8*g + r)*64 + (pbase + 1)*16 + l15] = s1[r] * 0.125f;
  }
  __syncthreads();

  // ---- row softmax (f32) -> bf16 probabilities ----
  if (t < 64) {
    float mx = -3.0e38f;
    for (int j = 0; j < 64; ++j) mx = fmaxf(mx, Ss[t*64 + j]);
    float sum = 0.f;
    for (int j = 0; j < 64; ++j) { float e = __expf(Ss[t*64 + j] - mx); Ss[t*64 + j] = e; sum += e; }
    float inv = 1.0f / sum;
    for (int j = 0; j < 64; ++j) Ps[t*72 + j] = f2bf(Ss[t*64 + j] * inv);
  }
  __syncthreads();

  // ---- out = P @ V  (B^T = V^T stored in Vts) ----
  v8f o0 = v8f_zero(), o1 = v8f_zero();
  #pragma unroll
  for (int s = 0; s < 2; ++s) {
    v16bf af = cat16(ld8(&Ps[qrow*72 + s*32 + 16*g]),
                     ld8(&Ps[qrow*72 + s*32 + 16*g + 8]));
    int d0r = (pbase + 0)*16 + l15;
    int d1r = (pbase + 1)*16 + l15;
    v16bf b0 = cat16(ld8(&Vts[d0r*72 + s*32 + 16*g]), ld8(&Vts[d0r*72 + s*32 + 16*g + 8]));
    v16bf b1 = cat16(ld8(&Vts[d1r*72 + s*32 + 16*g]), ld8(&Vts[d1r*72 + s*32 + 16*g + 8]));
    o0 = WMMA_BF16(af, b0, o0);
    o1 = WMMA_BF16(af, b1, o1);
  }
  #pragma unroll
  for (int r = 0; r < 8; ++r) {
    size_t mrow = (size_t)(b*64 + ntile*16 + 8*g + r) * DD + hoff;
    O[mrow + (pbase + 0)*16 + l15] = f2bf(o0[r]);
    O[mrow + (pbase + 1)*16 + l15] = f2bf(o1[r]);
  }
}

// ---------------------------------------------------------------------------
// LayerNorm: out = LN(base + add)*g + b ; optional f32 and bf16 outputs.
// grid = 16384 tokens, block = 256 (2 elems/thread over D=512)
// ---------------------------------------------------------------------------
__global__ __launch_bounds__(256) void ln_kernel(
    const float* __restrict__ base, const float* __restrict__ add,
    const float* __restrict__ gw, const float* __restrict__ bw,
    float* __restrict__ outf, bf16_t* __restrict__ outb)
{
  __shared__ float red[256];
  const int t = threadIdx.x;
  const size_t roff = (size_t)blockIdx.x * DD;
  float v0 = base[roff + t]       + add[roff + t];
  float v1 = base[roff + 256 + t] + add[roff + 256 + t];
  red[t] = v0 + v1;
  __syncthreads();
  for (int o = 128; o > 0; o >>= 1) { if (t < o) red[t] += red[t + o]; __syncthreads(); }
  float mean = red[0] * (1.0f / 512.0f);
  __syncthreads();
  float d0 = v0 - mean, d1 = v1 - mean;
  red[t] = d0*d0 + d1*d1;
  __syncthreads();
  for (int o = 128; o > 0; o >>= 1) { if (t < o) red[t] += red[t + o]; __syncthreads(); }
  float rstd = rsqrtf(red[0] * (1.0f / 512.0f) + 1e-5f);
  float o0 = d0 * rstd * gw[t]       + bw[t];
  float o1 = d1 * rstd * gw[256 + t] + bw[256 + t];
  if (outf) { outf[roff + t] = o0; outf[roff + 256 + t] = o1; }
  if (outb) { outb[roff + t] = f2bf(o0); outb[roff + 256 + t] = f2bf(o1); }
}

// ---------------------------------------------------------------------------
// Elementwise / layout kernels
// ---------------------------------------------------------------------------
__global__ void cvt_xc(const float* __restrict__ x, const float* __restrict__ x2,
                       bf16_t* __restrict__ XC) {
  int idx = blockIdx.x * 256 + threadIdx.x;        // MTOK*1024
  int m = idx >> 10, c = idx & 1023;
  float v = (c < DD) ? x[(size_t)m*DD + c] : x2[(size_t)m*DD + (c - DD)];
  XC[idx] = f2bf(v);
}
__global__ void wt_tr(const float* __restrict__ W, bf16_t* __restrict__ Wt, int K, int N) {
  int idx = blockIdx.x * 256 + threadIdx.x;        // N*K ; Wt[n,k] = W[k,n]
  int n = idx / K, k = idx - n*K;
  Wt[idx] = f2bf(W[(size_t)k * N + n]);
}
__global__ void conv_wt(const float* __restrict__ W, bf16_t* __restrict__ Wt) {
  int idx = blockIdx.x * 256 + threadIdx.x;        // 512*4608
  int o = idx / 4608, r = idx - o*4608;
  int tap = r >> 9, c = r & 511;                   // Wt[o][tap*512+c] = W[o][c][ky][kx]
  Wt[idx] = f2bf(W[((size_t)o * 512 + c) * 9 + tap]);
}
// raw-view permutation:  xf[b,p,c] = mix[b, c/8, (c%8)*64 + p]
__global__ void perm_mix(const float* __restrict__ mix, bf16_t* __restrict__ P) {
  int idx = blockIdx.x * 256 + threadIdx.x;        // MTOK*512
  int tok = idx >> 9, c = idx & 511;
  int b = tok >> 6, p = tok & 63;
  P[idx] = f2bf(mix[((size_t)(b*64 + (c >> 3)))*DD + ((c & 7) << 6) + p]);
}
__global__ void mean_batch(const float* __restrict__ mix, float* __restrict__ xmean) {
  int idx = blockIdx.x * 256 + threadIdx.x;        // 64*512
  int p = idx >> 9, c = idx & 511;
  size_t base = ((size_t)(c >> 3)) * DD + ((c & 7) << 6) + p;
  float s = 0.f;
  for (int b = 0; b < 256; ++b) s += mix[base + (size_t)b * 64 * DD];
  xmean[idx] = s * (1.0f / 256.0f);
}
// S = xmean + gamma*attn_out + xf   (conv input, token-major bf16)
__global__ void build_s(const float* __restrict__ mix, const bf16_t* __restrict__ oc,
                        const float* __restrict__ xmean, const float* __restrict__ gamma,
                        bf16_t* __restrict__ S) {
  int idx = blockIdx.x * 256 + threadIdx.x;        // MTOK*512
  int tok = idx >> 9, c = idx & 511;
  int b = tok >> 6, p = tok & 63;
  float xi  = mix[((size_t)(b*64 + (c >> 3)))*DD + ((c & 7) << 6) + p];
  float att = gamma[0] * bf2f(oc[idx]) + xi;
  S[idx] = f2bf(xmean[(p << 9) + c] + att);
}

// ---------------------------------------------------------------------------
extern "C" void kernel_launch(void* const* d_in, const int* in_sizes, int n_in,
                              void* d_out, int out_size, void* d_ws, size_t ws_size,
                              hipStream_t stream) {
  (void)in_sizes; (void)n_in; (void)out_size; (void)ws_size;
  const float* x        = (const float*)d_in[0];
  const float* x2       = (const float*)d_in[1];
  const float* W_mix_q  = (const float*)d_in[2];
  const float* b_mix_q  = (const float*)d_in[3];
  const float* W_mix_qp = (const float*)d_in[4];
  const float* b_mix_qp = (const float*)d_in[5];
  const float* Wq_ca    = (const float*)d_in[6];
  const float* bq_ca    = (const float*)d_in[7];
  const float* Wk_ca    = (const float*)d_in[8];
  const float* bk_ca    = (const float*)d_in[9];
  const float* Wv_ca    = (const float*)d_in[10];
  const float* bv_ca    = (const float*)d_in[11];
  const float* gamma    = (const float*)d_in[12];
  const float* W_conv3  = (const float*)d_in[13];
  const float* b_conv3  = (const float*)d_in[14];
  const float* W_rgb_k  = (const float*)d_in[15];
  const float* b_rgb_k  = (const float*)d_in[16];
  const float* W_rgb_v  = (const float*)d_in[17];
  const float* b_rgb_v  = (const float*)d_in[18];
  const float* W_rgb_o  = (const float*)d_in[19];
  const float* b_rgb_o  = (const float*)d_in[20];
  const float* W_inf_k  = (const float*)d_in[21];
  const float* b_inf_k  = (const float*)d_in[22];
  const float* W_inf_v  = (const float*)d_in[23];
  const float* b_inf_v  = (const float*)d_in[24];
  const float* W_inf_o  = (const float*)d_in[25];
  const float* b_inf_o  = (const float*)d_in[26];
  const float* g_in_rgb  = (const float*)d_in[27];
  const float* be_in_rgb = (const float*)d_in[28];
  const float* g_in_inf  = (const float*)d_in[29];
  const float* be_in_inf = (const float*)d_in[30];
  const float* g_out_rgb  = (const float*)d_in[31];
  const float* be_out_rgb = (const float*)d_in[32];
  const float* g_out_inf  = (const float*)d_in[33];
  const float* be_out_inf = (const float*)d_in[34];
  const float* W_m_rgb1 = (const float*)d_in[35];
  const float* b_m_rgb1 = (const float*)d_in[36];
  const float* W_m_rgb2 = (const float*)d_in[37];
  const float* b_m_rgb2 = (const float*)d_in[38];
  const float* W_m_inf1 = (const float*)d_in[39];
  const float* b_m_inf1 = (const float*)d_in[40];
  const float* W_m_inf2 = (const float*)d_in[41];
  const float* b_m_inf2 = (const float*)d_in[42];
  float* outp = (float*)d_out;

  char* wsb = (char*)d_ws;
  size_t off = 0;
  auto alloc = [&](size_t bytes) -> void* {
    void* p = wsb + off;
    off = (off + bytes + 255) & ~(size_t)255;
    return p;
  };
  const size_t SZ_BF = (size_t)MTOK * DD * 2;      // one [M,512] bf16 buffer
  const size_t SZ_F  = (size_t)MTOK * DD * 4;      // one [M,512] f32 buffer

  bf16_t* XC       = (bf16_t*)alloc((size_t)MTOK * 1024 * 2);
  bf16_t* Wt_mixq  = (bf16_t*)alloc(512 * 1024 * 2);
  bf16_t* Wt_qca   = (bf16_t*)alloc(512 * 512 * 2);
  bf16_t* Wt_kca   = (bf16_t*)alloc(512 * 512 * 2);
  bf16_t* Wt_vca   = (bf16_t*)alloc(512 * 512 * 2);
  bf16_t* Wt_mixqp = (bf16_t*)alloc(512 * 512 * 2);
  bf16_t* Wt_rgbk  = (bf16_t*)alloc(512 * 512 * 2);
  bf16_t* Wt_rgbv  = (bf16_t*)alloc(512 * 512 * 2);
  bf16_t* Wt_rgbo  = (bf16_t*)alloc(512 * 512 * 2);
  bf16_t* Wt_infk  = (bf16_t*)alloc(512 * 512 * 2);
  bf16_t* Wt_infv  = (bf16_t*)alloc(512 * 512 * 2);
  bf16_t* Wt_info  = (bf16_t*)alloc(512 * 512 * 2);
  bf16_t* Wt_conv  = (bf16_t*)alloc((size_t)512 * 4608 * 2);
  bf16_t* Wt_m1r   = (bf16_t*)alloc((size_t)2048 * 512 * 2);
  bf16_t* Wt_m2r   = (bf16_t*)alloc((size_t)512 * 2048 * 2);
  bf16_t* Wt_m1i   = (bf16_t*)alloc((size_t)2048 * 512 * 2);
  bf16_t* Wt_m2i   = (bf16_t*)alloc((size_t)512 * 2048 * 2);
  float*  mixf     = (float*)alloc(SZ_F);          // later reused as 2x bf16 attn-out
  bf16_t* BF0      = (bf16_t*)alloc(SZ_BF);        // P   -> q
  bf16_t* BF1      = (bf16_t*)alloc(SZ_BF);        // qc  -> rgb_k
  bf16_t* BF2      = (bf16_t*)alloc(SZ_BF);        // kc  -> rgb_v
  bf16_t* BF3      = (bf16_t*)alloc(SZ_BF);        // vc  -> inf_k
  bf16_t* BF4      = (bf16_t*)alloc(SZ_BF);        // oc  -> inf_v
  bf16_t* BF5      = (bf16_t*)alloc(SZ_BF);        // S (conv input)
  bf16_t* mix2     = (bf16_t*)alloc(SZ_BF);
  float*  xmean    = (float*)alloc(64 * 512 * 4);
  float*  cax      = (float*)alloc(SZ_F);          // reused for MLP2 outputs
  float*  cax2     = (float*)alloc(SZ_F);
  float*  xlnf_r   = (float*)alloc(SZ_F);
  float*  xlnf_i   = (float*)alloc(SZ_F);
  bf16_t* xlnb_r   = (bf16_t*)alloc(SZ_BF);
  bf16_t* xlnb_i   = (bf16_t*)alloc(SZ_BF);
  bf16_t* hidden   = (bf16_t*)alloc((size_t)MTOK * 2048 * 2);
  bf16_t* rgb_ao   = (bf16_t*)mixf;                // reuse (same total bytes)
  bf16_t* inf_ao   = (bf16_t*)mixf + (size_t)MTOK * DD;

  const dim3 blk(256);
  const dim3 gN512(4, MTOK / 128);
  const dim3 gN2048(16, MTOK / 128);

  // ---- phase 0: precision conversion + weight pre-transposition ----
  cvt_xc<<<(MTOK * 1024) / 256, blk, 0, stream>>>(x, x2, XC);
  wt_tr<<<(512 * 1024) / 256, blk, 0, stream>>>(W_mix_q, Wt_mixq, 1024, 512);
  wt_tr<<<(512 * 512) / 256, blk, 0, stream>>>(Wq_ca,    Wt_qca,   512, 512);
  wt_tr<<<(512 * 512) / 256, blk, 0, stream>>>(Wk_ca,    Wt_kca,   512, 512);
  wt_tr<<<(512 * 512) / 256, blk, 0, stream>>>(Wv_ca,    Wt_vca,   512, 512);
  wt_tr<<<(512 * 512) / 256, blk, 0, stream>>>(W_mix_qp, Wt_mixqp, 512, 512);
  wt_tr<<<(512 * 512) / 256, blk, 0, stream>>>(W_rgb_k,  Wt_rgbk,  512, 512);
  wt_tr<<<(512 * 512) / 256, blk, 0, stream>>>(W_rgb_v,  Wt_rgbv,  512, 512);
  wt_tr<<<(512 * 512) / 256, blk, 0, stream>>>(W_rgb_o,  Wt_rgbo,  512, 512);
  wt_tr<<<(512 * 512) / 256, blk, 0, stream>>>(W_inf_k,  Wt_infk,  512, 512);
  wt_tr<<<(512 * 512) / 256, blk, 0, stream>>>(W_inf_v,  Wt_infv,  512, 512);
  wt_tr<<<(512 * 512) / 256, blk, 0, stream>>>(W_inf_o,  Wt_info,  512, 512);
  wt_tr<<<(512 * 2048) / 256, blk, 0, stream>>>(W_m_rgb1, Wt_m1r, 512, 2048);
  wt_tr<<<(2048 * 512) / 256, blk, 0, stream>>>(W_m_rgb2, Wt_m2r, 2048, 512);
  wt_tr<<<(512 * 2048) / 256, blk, 0, stream>>>(W_m_inf1, Wt_m1i, 512, 2048);
  wt_tr<<<(2048 * 512) / 256, blk, 0, stream>>>(W_m_inf2, Wt_m2i, 2048, 512);
  conv_wt<<<(512 * 4608) / 256, blk, 0, stream>>>(W_conv3, Wt_conv);

  // ---- phase 1: mix = [x|x2] @ W_mix_q ; permuted raw-view ; complex attn ----
  gemm_nt<<<gN512, blk, 0, stream>>>(XC, 1024, Wt_mixq, b_mix_q, mixf, 512, 1024, 0);
  perm_mix<<<(MTOK * 512) / 256, blk, 0, stream>>>(mixf, BF0);
  gemm_nt<<<gN512, blk, 0, stream>>>(BF0, 512, Wt_qca, bq_ca, BF1, 512, 512, 1);
  gemm_nt<<<gN512, blk, 0, stream>>>(BF0, 512, Wt_kca, bk_ca, BF2, 512, 512, 1);
  gemm_nt<<<gN512, blk, 0, stream>>>(BF0, 512, Wt_vca, bv_ca, BF3, 512, 512, 1);
  attn_kernel<<<2048, blk, 0, stream>>>(BF1, BF2, BF3, BF4);
  mean_batch<<<128, blk, 0, stream>>>(mixf, xmean);
  build_s<<<(MTOK * 512) / 256, blk, 0, stream>>>(mixf, BF4, xmean, gamma, BF5);

  // ---- phase 2: 3x3 conv as implicit-im2col GEMM -> mix2 ----
  conv_gemm<<<gN512, blk, 0, stream>>>(BF5, Wt_conv, b_conv3, mix2);

  // ---- phase 3: cross attention ----
  gemm_nt<<<gN512, blk, 0, stream>>>(mix2,     512, Wt_mixqp, b_mix_qp, BF0, 512, 512, 1);
  gemm_nt<<<gN512, blk, 0, stream>>>(XC,      1024, Wt_rgbk,  b_rgb_k,  BF1, 512, 512, 1);
  gemm_nt<<<gN512, blk, 0, stream>>>(XC,      1024, Wt_rgbv,  b_rgb_v,  BF2, 512, 512, 1);
  gemm_nt<<<gN512, blk, 0, stream>>>(XC + 512, 1024, Wt_infk, b_inf_k,  BF3, 512, 512, 1);
  gemm_nt<<<gN512, blk, 0, stream>>>(XC + 512, 1024, Wt_infv, b_inf_v,  BF4, 512, 512, 1);
  attn_kernel<<<2048, blk, 0, stream>>>(BF0, BF3, BF2, rgb_ao);  // q, inf_k, rgb_v
  attn_kernel<<<2048, blk, 0, stream>>>(BF0, BF1, BF4, inf_ao);  // q, rgb_k, inf_v
  gemm_nt<<<gN512, blk, 0, stream>>>(rgb_ao, 512, Wt_rgbo, b_rgb_o, cax,  512, 512, 0);
  gemm_nt<<<gN512, blk, 0, stream>>>(inf_ao, 512, Wt_info, b_inf_o, cax2, 512, 512, 0);

  // ---- phase 4: LN1 + MLP + LN2, both branches ----
  ln_kernel<<<MTOK, blk, 0, stream>>>(x,  cax,  g_in_rgb, be_in_rgb, xlnf_r, xlnb_r);
  ln_kernel<<<MTOK, blk, 0, stream>>>(x2, cax2, g_in_inf, be_in_inf, xlnf_i, xlnb_i);

  gemm_nt<<<gN2048, blk, 0, stream>>>(xlnb_r, 512,  Wt_m1r, b_m_rgb1, hidden, 2048, 512,  3);
  gemm_nt<<<gN512,  blk, 0, stream>>>(hidden, 2048, Wt_m2r, b_m_rgb2, cax,    512,  2048, 0);
  ln_kernel<<<MTOK, blk, 0, stream>>>(xlnf_r, cax, g_out_rgb, be_out_rgb, outp, nullptr);

  gemm_nt<<<gN2048, blk, 0, stream>>>(xlnb_i, 512,  Wt_m1i, b_m_inf1, hidden, 2048, 512,  3);
  gemm_nt<<<gN512,  blk, 0, stream>>>(hidden, 2048, Wt_m2i, b_m_inf2, cax,    512,  2048, 0);
  ln_kernel<<<MTOK, blk, 0, stream>>>(xlnf_i, cax, g_out_inf, be_out_inf,
                                      outp + (size_t)MTOK * DD, nullptr);
}
